// TriplaneRendererFG_39900246179913
// MI455X (gfx1250) — matrix operand order repaired and checked by reading, not compiled, not run on hardware
//
#include <hip/hip_runtime.h>
#include <cstdint>

// ---------------------------------------------------------------------------
// TriplaneRendererFG for MI455X (gfx1250, wave32, WMMA f32_16x16x32_f16)
//
//   k_z        : z = z_slots + slot_pos.xy @ pos_w + pos_b            (8x64)
//   k_fmap     : fmap[n,y,x,c] = z + PE(grid)                          f16
//   k_conv<64,64,relu> x2 : 3x3 conv as implicit GEMM (WMMA),
//                           activations staged to LDS via async-to-LDS b128
//   k_conv<64,96,linear>  : conv -> triplane (ch = plane*32+c)         f16
//   k_decode   : bilinear triplane sample -> 96 feat -> MLP (WMMA)
//   k_combine  : cross-slot mask normalization
//
// WMMA fragments are two aligned ds_load_b128s. B operands live [n][k]
// (this code uses contiguous K order, k = 16h + s). A operands are stored
// unswizzled, in the same contiguous K order as B (aswz() is identity);
// the async-copy destinations keep the global channel order.
// Conv waves own 32 pixels (2 M-tiles) so every B fragment feeds 2 WMMAs,
// and B fragments are preloaded into registers before each WMMA burst so
// s_wait_dscnt waits are graduated instead of full drains.
// ---------------------------------------------------------------------------

typedef _Float16 h16;
typedef __attribute__((ext_vector_type(16))) _Float16 v16h;
typedef __attribute__((ext_vector_type(8)))  float    v8f;
typedef __attribute__((ext_vector_type(4)))  unsigned int u32x4;

#define RES   128
#define NSLOT 8
#define MPTS  65536
#define ZDIM  64
#define TRIC  96

// d_out float layout (total 4,980,736):
#define OUT_MASKED   262144u
#define OUT_UNMASK   2359296u
#define OUT_MASKS    4456448u

// Position of logical k within A-operand storage: identity (contiguous K)
__device__ __forceinline__ int aswz(int k) {
  return k;
}

// Load 16 f16 (32B, 32B-aligned) as one WMMA operand: 2x ds_load_b128.
__device__ __forceinline__ v16h load_frag(const h16* p) {
  union { v16h v; u32x4 q[2]; } f;
  const u32x4* q = (const u32x4*)p;
  f.q[0] = q[0];
  f.q[1] = q[1];
  return f.v;
}
__device__ __forceinline__ v8f wmma16(v16h a, v16h b, v8f c) {
  return __builtin_amdgcn_wmma_f32_16x16x32_f16(false, a, false, b, (short)0, c,
                                                false, false);
}

// ---------------- k_z ------------------------------------------------------
__global__ void k_z(const float* __restrict__ z_slots,
                    const float* __restrict__ slot_pos,
                    const float* __restrict__ pos_w,
                    const float* __restrict__ pos_b,
                    float* __restrict__ zbuf) {
  int i = threadIdx.x;
  if (i < NSLOT * ZDIM) {
    int n = i / ZDIM, c = i % ZDIM;
    zbuf[i] = z_slots[i] + slot_pos[n * 3 + 0] * pos_w[c] +
              slot_pos[n * 3 + 1] * pos_w[ZDIM + c] + pos_b[c];
  }
}

// ---------------- k_fmap: 8 channels/thread, one b128 store ----------------
__global__ void __launch_bounds__(256)
k_fmap(const float* __restrict__ zbuf, const float* __restrict__ pe_w,
       const float* __restrict__ pe_b, h16* __restrict__ fmap) {
  int idx = blockIdx.x * 256 + threadIdx.x;  // N*H*W*8 = 1048576 exact
  int g = idx & 7;
  int rest = idx >> 3;
  int x = rest & 127; rest >>= 7;
  int y = rest & 127;
  int n = rest >> 7;
  float xx = -1.f + (2.f * x + 1.f) / 128.f;
  float yy = -1.f + (2.f * y + 1.f) / 128.f;
  union { h16 h[8]; u32x4 q; } o;
#pragma unroll
  for (int j = 0; j < 8; ++j) {
    int c = g * 8 + j;
    float v = zbuf[n * ZDIM + c] + xx * pe_w[c] + yy * pe_w[64 + c] -
              xx * pe_w[128 + c] - yy * pe_w[192 + c] + pe_b[c];
    o.h[j] = (h16)v;
  }
  ((u32x4*)fmap)[idx] = o.q;
}

// ---------------- k_conv: 3x3 SAME conv as implicit GEMM via WMMA ----------
// Grid: N*64 blocks, 256 threads (8 waves). Block covers rows y0, y0+1.
// Wave wid: row y0 + (wid>>2), 32 pixels at x = (wid&3)*32 (2 M-tiles).
// LDS: alds[4][130][IC] f16 (contiguous K), wlds[9][OC][IC] f16.
template <int IC, int OC, bool RELU>
__global__ void __launch_bounds__(256)
k_conv(const h16* __restrict__ in, const float* __restrict__ w,
       const float* __restrict__ bias, h16* __restrict__ out) {
  extern __shared__ char smem[];
  h16* alds = (h16*)smem;              // 4*130*IC, channels in contiguous order
  h16* wlds = alds + 4 * 130 * IC;     // 9*OC*IC, [tap][oc][ic]
  const int tid = threadIdx.x;
  const int slot = blockIdx.x >> 6;
  const int y0 = (blockIdx.x & 63) * 2;
  const int G = IC / 8;                // 16B channel groups per cell (8)

  // ---- zero-fill halo columns (xc=0,129) and out-of-range rows ----
  u32x4 zq = {0u, 0u, 0u, 0u};
  u32x4* aq4 = (u32x4*)alds;
  for (int i = tid; i < 2 * 4 * G; i += 256) {
    int g = i % G; int rest = i / G;
    int r = rest % 4; int xc = (rest / 4) ? 129 : 0;
    aq4[(r * 130 + xc) * G + g] = zq;
  }
  if (y0 == 0) {
    for (int i = tid; i < 130 * G; i += 256)
      aq4[(0 * 130 + (i / G)) * G + (i % G)] = zq;
  }
  if (y0 == 126) {
    for (int i = tid; i < 130 * G; i += 256)
      aq4[(3 * 130 + (i / G)) * G + (i % G)] = zq;
  }

  // ---- async global->LDS copy of valid cells, unswizzled, at 16B granularity -
  {
    const uint64_t gbase = (uint64_t)in;
    for (int i = tid; i < 4 * 128 * G; i += 256) {
      int g = i % G; int rest = i / G;
      int xc1 = rest & 127;            // x = xc1, LDS column xc1+1
      int r = rest >> 7;               // 0..3 -> yy = y0 + r - 1
      int yy = y0 + r - 1;
      if (yy >= 0 && yy < RES) {
        uint32_t goff =
            (uint32_t)((((slot * RES + yy) * RES + xc1) * IC + g * 8) * 2);
        int gd = g;  // identity (contiguous K)
        uint32_t laddr = (uint32_t)(uintptr_t)
            &alds[((r * 130) + (xc1 + 1)) * IC + gd * 8];
        asm volatile("global_load_async_to_lds_b128 %0, %1, %2"
                     :: "v"(laddr), "v"(goff), "s"(gbase) : "memory");
      }
    }
  }

  // ---- stage weights f32 OIHW -> f16 [tap][oc][ic] ----
  for (int i = tid; i < 9 * OC * IC; i += 256) {
    int ic = i % IC;
    int rest = i / IC;
    int oc = rest % OC;
    int t = rest / OC;
    int ky = t / 3, kx = t % 3;
    wlds[i] = (h16)w[((oc * IC + ic) * 3 + ky) * 3 + kx];
  }
  asm volatile("s_wait_asynccnt 0" ::: "memory");
  __syncthreads();

  const int lane = tid & 31, wid = tid >> 5;
  const int khalf = lane >> 4, m = lane & 15;
  const int ry = wid >> 2;             // row within block (0/1)
  const int y = y0 + ry;
  const int xBase = (wid & 3) * 32;    // 32 pixels per wave

  v8f zero = {0.f, 0.f, 0.f, 0.f, 0.f, 0.f, 0.f, 0.f};
  v8f acc0[OC / 16], acc1[OC / 16];
#pragma unroll
  for (int ot = 0; ot < OC / 16; ++ot) { acc0[ot] = zero; acc1[ot] = zero; }

  for (int t = 0; t < 9; ++t) {
    const int ky = t / 3, kx = t % 3;
    const h16* acol = &alds[(((ry + ky) * 130) + (xBase + m + kx)) * IC];
#pragma unroll
    for (int kc = 0; kc < IC / 32; ++kc) {
      v16h a0 = load_frag(acol + kc * 32 + khalf * 16);
      v16h a1 = load_frag(acol + 16 * IC + kc * 32 + khalf * 16);
      v16h bf[OC / 16];
#pragma unroll
      for (int ot = 0; ot < OC / 16; ++ot)
        bf[ot] = load_frag(&wlds[((t * OC) + ot * 16 + m) * IC +
                                 kc * 32 + khalf * 16]);
#pragma unroll
      for (int ot = 0; ot < OC / 16; ++ot) {
        acc0[ot] = wmma16(a0, bf[ot], acc0[ot]);
        acc1[ot] = wmma16(a1, bf[ot], acc1[ot]);
      }
    }
  }

  // D tile: vgpr r -> pixel m = r + 8*khalf; col n = lane&15.
#pragma unroll
  for (int ot = 0; ot < OC / 16; ++ot) {
    int n = ot * 16 + m;
    float bv = bias[n];
#pragma unroll
    for (int r = 0; r < 8; ++r) {
      int px0 = xBase + r + 8 * khalf;
      float v0 = acc0[ot][r] + bv;
      float v1 = acc1[ot][r] + bv;
      if (RELU) { v0 = fmaxf(v0, 0.f); v1 = fmaxf(v1, 0.f); }
      out[((size_t)(slot * RES + y) * RES + px0) * OC + n] = (h16)v0;
      out[((size_t)(slot * RES + y) * RES + px0 + 16) * OC + n] = (h16)v1;
    }
  }
}

// ---------------- k_decode: sample triplane + MLP heads via WMMA -----------
__global__ void __launch_bounds__(256)
k_decode(const h16* __restrict__ tri, const float* __restrict__ sc,
         const float* __restrict__ sp, const float* __restrict__ dw0,
         const float* __restrict__ db0, const float* __restrict__ dw1,
         const float* __restrict__ db1, const float* __restrict__ dnw,
         const float* __restrict__ dnb, const float* __restrict__ c0w,
         const float* __restrict__ c0b, const float* __restrict__ c1w,
         const float* __restrict__ c1b, float* __restrict__ outp) {
  extern __shared__ char smem[];
  h16* Wd0 = (h16*)smem;        // 96x96  [n][k]
  h16* Wd1 = Wd0 + 9216;        // 96x96
  h16* Wde = Wd1 + 9216;        // 16x96  (n==0 density row)
  h16* Wc0 = Wde + 1536;        // 32x96  (n<24 valid)
  h16* Wc1 = Wc0 + 3072;        // 16x32  (n<3, k<24 valid)
  h16* astAll = Wc1 + 512;      // 8 waves x 16 pts x 96 (contiguous K)
  const int tid = threadIdx.x;

  for (int i = tid; i < 9216; i += 256) {
    int n = i / 96, k = i % 96;
    Wd0[i] = (h16)dw0[k * 96 + n];
    Wd1[i] = (h16)dw1[k * 96 + n];
  }
  for (int i = tid; i < 1536; i += 256) {
    int n = i / 96, k = i % 96;
    Wde[i] = (n == 0) ? (h16)dnw[k] : (h16)0.f;
  }
  for (int i = tid; i < 3072; i += 256) {
    int n = i / 96, k = i % 96;
    Wc0[i] = (n < 24) ? (h16)c0w[k * 24 + n] : (h16)0.f;
  }
  for (int i = tid; i < 512; i += 256) {
    int n = i / 32, k = i % 32;
    Wc1[i] = (n < 3 && k < 24) ? (h16)c1w[k * 3 + n] : (h16)0.f;
  }
  __syncthreads();

  const int lane = tid & 31, wid = tid >> 5;
  const int khalf = lane >> 4, m = lane & 15;
  h16* ast = astAll + wid * 16 * 96;

  const int tileId = blockIdx.x * 8 + wid;
  const int slot = tileId >> 12;
  const int ptBase = (tileId & 4095) * 16;

  // ---- bilinear triplane sampling into ast[m][aswz(ch)] ----
  {
    const int pt = ptBase + m;
    const float cx = sc[(slot * MPTS + pt) * 3 + 0] - sp[slot * 3 + 0];
    const float cy = sc[(slot * MPTS + pt) * 3 + 1] - sp[slot * 3 + 1];
    const float cz = sc[(slot * MPTS + pt) * 3 + 2] - sp[slot * 3 + 2];
#pragma unroll
    for (int p = 0; p < 3; ++p) {
      // projections: p0=(x,y) p1=(x,z) p2=(z,x)
      float u = (p == 2) ? cz : cx;
      float v = (p == 0) ? cy : ((p == 1) ? cz : cx);
      float fx = (u + 1.f) * 64.f - 0.5f;
      float fy = (v + 1.f) * 64.f - 0.5f;
      float x0f = floorf(fx), y0f = floorf(fy);
      float wx = fx - x0f, wy = fy - y0f;
      int ix0 = (int)x0f, iy0 = (int)y0f;
      const int chBase = p * 32 + khalf * 16;
      float facc[16];
#pragma unroll
      for (int c = 0; c < 16; ++c) facc[c] = 0.f;
#pragma unroll
      for (int corner = 0; corner < 4; ++corner) {
        int ix = ix0 + (corner & 1);
        int iy = iy0 + (corner >> 1);
        float wgt = ((corner & 1) ? wx : 1.f - wx) *
                    ((corner >> 1) ? wy : 1.f - wy);
        if (ix >= 0 && ix < RES && iy >= 0 && iy < RES) {
          const h16* s =
              &tri[((size_t)(slot * RES + iy) * RES + ix) * TRIC + chBase];
#pragma unroll
          for (int c = 0; c < 16; ++c) facc[c] += wgt * (float)s[c];
        }
      }
#pragma unroll
      for (int c = 0; c < 16; ++c)
        ast[m * 96 + aswz(chBase + c)] = (h16)facc[c];
    }
  }
  __syncthreads();

  v8f zero = {0.f, 0.f, 0.f, 0.f, 0.f, 0.f, 0.f, 0.f};
  const h16* arow = &ast[m * 96];
  v8f acc[6];

  // ---- h1 = relu(feat @ dec_w0 + b0) ----
#pragma unroll
  for (int ot = 0; ot < 6; ++ot) acc[ot] = zero;
#pragma unroll
  for (int kc = 0; kc < 3; ++kc) {
    v16h a = load_frag(arow + kc * 32 + khalf * 16);
    v16h bf[6];
#pragma unroll
    for (int ot = 0; ot < 6; ++ot)
      bf[ot] = load_frag(&Wd0[(ot * 16 + m) * 96 + kc * 32 + khalf * 16]);
#pragma unroll
    for (int ot = 0; ot < 6; ++ot) acc[ot] = wmma16(a, bf[ot], acc[ot]);
  }
  __syncthreads();
#pragma unroll
  for (int ot = 0; ot < 6; ++ot) {
    int ns = aswz(ot * 16 + m);
    float bv = db0[ot * 16 + m];
#pragma unroll
    for (int r = 0; r < 8; ++r)
      ast[(r + 8 * khalf) * 96 + ns] = (h16)fmaxf(acc[ot][r] + bv, 0.f);
  }
  __syncthreads();

  // ---- h2 = relu(h1 @ dec_w1 + b1) ----
#pragma unroll
  for (int ot = 0; ot < 6; ++ot) acc[ot] = zero;
#pragma unroll
  for (int kc = 0; kc < 3; ++kc) {
    v16h a = load_frag(arow + kc * 32 + khalf * 16);
    v16h bf[6];
#pragma unroll
    for (int ot = 0; ot < 6; ++ot)
      bf[ot] = load_frag(&Wd1[(ot * 16 + m) * 96 + kc * 32 + khalf * 16]);
#pragma unroll
    for (int ot = 0; ot < 6; ++ot) acc[ot] = wmma16(a, bf[ot], acc[ot]);
  }
  __syncthreads();
#pragma unroll
  for (int ot = 0; ot < 6; ++ot) {
    int ns = aswz(ot * 16 + m);
    float bv = db1[ot * 16 + m];
#pragma unroll
    for (int r = 0; r < 8; ++r)
      ast[(r + 8 * khalf) * 96 + ns] = (h16)fmaxf(acc[ot][r] + bv, 0.f);
  }
  __syncthreads();

  // ---- density (h2 @ dens_w) and c0 = relu(h2 @ col_w0 + b) ----
  v8f accd = zero, accc0 = zero, accc1 = zero;
#pragma unroll
  for (int kc = 0; kc < 3; ++kc) {
    v16h a = load_frag(arow + kc * 32 + khalf * 16);
    v16h bd = load_frag(&Wde[m * 96 + kc * 32 + khalf * 16]);
    v16h b0 = load_frag(&Wc0[m * 96 + kc * 32 + khalf * 16]);
    v16h b1 = load_frag(&Wc0[(16 + m) * 96 + kc * 32 + khalf * 16]);
    accd = wmma16(a, bd, accd);
    accc0 = wmma16(a, b0, accc0);
    accc1 = wmma16(a, b1, accc1);
  }
  __syncthreads();
  {
    int n0 = m, n1 = 16 + m;
    int ns0 = aswz(n0), ns1 = aswz(n1);
    float bv0 = (n0 < 24) ? c0b[n0] : 0.f;
    float bv1 = (n1 < 24) ? c0b[n1] : 0.f;
#pragma unroll
    for (int r = 0; r < 8; ++r) {
      ast[(r + 8 * khalf) * 96 + ns0] = (h16)fmaxf(accc0[r] + bv0, 0.f);
      ast[(r + 8 * khalf) * 96 + ns1] = (h16)fmaxf(accc1[r] + bv1, 0.f);
    }
  }
  __syncthreads();

  // ---- color = c0 @ col_w1 (cols 0..2) ----
  v8f accr = zero;
  {
    v16h a = load_frag(arow + khalf * 16);
    v16h b = load_frag(&Wc1[m * 32 + khalf * 16]);
    accr = wmma16(a, b, accr);
  }

  // ---- write unmasked_raws = [ (tanh(color)+1)/2 , relu(density) ] ----
  if (m < 3) {
#pragma unroll
    for (int r = 0; r < 8; ++r) {
      int pt = ptBase + r + 8 * khalf;
      float col = accr[r] + c1b[m];
      outp[OUT_UNMASK + ((size_t)(slot * MPTS + pt)) * 4 + m] =
          (tanhf(col) + 1.f) * 0.5f;
    }
  }
  if (m == 0) {
#pragma unroll
    for (int r = 0; r < 8; ++r) {
      int pt = ptBase + r + 8 * khalf;
      float ax = fabsf(sc[(slot * MPTS + pt) * 3 + 0] - sp[slot * 3 + 0]);
      float ay = fabsf(sc[(slot * MPTS + pt) * 3 + 1] - sp[slot * 3 + 1]);
      float az = fabsf(sc[(slot * MPTS + pt) * 3 + 2] - sp[slot * 3 + 2]);
      float d = accd[r] + dnb[0];
      if (ax > 1.f || ay > 1.f || az > 1.f) d = 0.f;
      outp[OUT_UNMASK + ((size_t)(slot * MPTS + pt)) * 4 + 3] = fmaxf(d, 0.f);
    }
  }
}

// ---------------- k_combine: cross-slot mask normalization -----------------
__global__ void __launch_bounds__(256)
k_combine(float* __restrict__ outp) {
  int pt = blockIdx.x * 256 + threadIdx.x;  // 65536 exact
  float u[NSLOT][4];
  float msum = 0.f;
#pragma unroll
  for (int n = 0; n < NSLOT; ++n) {
#pragma unroll
    for (int c = 0; c < 4; ++c)
      u[n][c] = outp[OUT_UNMASK + ((size_t)(n * MPTS + pt)) * 4 + c];
    msum += u[n][3];
  }
  float inv = 1.f / (msum + 1e-8f);
  float raws[4] = {0.f, 0.f, 0.f, 0.f};
#pragma unroll
  for (int n = 0; n < NSLOT; ++n) {
    float mk = u[n][3] * inv;
    outp[OUT_MASKS + (size_t)n * MPTS + pt] = mk;
#pragma unroll
    for (int c = 0; c < 4; ++c) {
      float mv = u[n][c] * mk;
      outp[OUT_MASKED + ((size_t)(n * MPTS + pt)) * 4 + c] = mv;
      raws[c] += mv;
    }
  }
#pragma unroll
  for (int c = 0; c < 4; ++c) outp[(size_t)pt * 4 + c] = raws[c];
}

// ---------------------------------------------------------------------------
extern "C" void kernel_launch(void* const* d_in, const int* in_sizes, int n_in,
                              void* d_out, int out_size, void* d_ws,
                              size_t ws_size, hipStream_t stream) {
  const float* sc    = (const float*)d_in[0];
  const float* zs    = (const float*)d_in[1];
  const float* sp    = (const float*)d_in[2];
  const float* pos_w = (const float*)d_in[3];
  const float* pos_b = (const float*)d_in[4];
  const float* pe_w  = (const float*)d_in[5];
  const float* pe_b  = (const float*)d_in[6];
  const float* c1_w  = (const float*)d_in[7];
  const float* c1_b  = (const float*)d_in[8];
  const float* c2_w  = (const float*)d_in[9];
  const float* c2_b  = (const float*)d_in[10];
  const float* c3_w  = (const float*)d_in[11];
  const float* c3_b  = (const float*)d_in[12];
  const float* dw0   = (const float*)d_in[13];
  const float* db0   = (const float*)d_in[14];
  const float* dw1   = (const float*)d_in[15];
  const float* db1   = (const float*)d_in[16];
  const float* dnw   = (const float*)d_in[17];
  const float* dnb   = (const float*)d_in[18];
  const float* c0w   = (const float*)d_in[19];
  const float* c0b   = (const float*)d_in[20];
  const float* c1wc  = (const float*)d_in[21];
  const float* c1bc  = (const float*)d_in[22];
  (void)in_sizes; (void)n_in; (void)out_size; (void)ws_size;

  char* base = (char*)d_ws;
  float* zbuf = (float*)base;                       // 512 f32
  h16* bufA = (h16*)(base + 4096);                  // 8*128*128*64 f16
  h16* bufB = (h16*)(base + 4096 + 16777216);       // 8*128*128*64 f16
  h16* bufT = (h16*)(base + 4096 + 2 * 16777216);   // 8*128*128*96 f16

  const size_t lds_conv64 = (size_t)(4 * 130 * 64 + 9 * 64 * 64) * 2;  // 140288
  const size_t lds_conv96 = (size_t)(4 * 130 * 64 + 9 * 96 * 64) * 2;  // 177152
  const size_t lds_dec =
      (size_t)(9216 * 2 + 1536 + 3072 + 512 + 8 * 16 * 96) * 2;        //  71680

  (void)hipFuncSetAttribute((const void*)&k_conv<64, 64, true>,
                            hipFuncAttributeMaxDynamicSharedMemorySize,
                            (int)lds_conv64);
  (void)hipFuncSetAttribute((const void*)&k_conv<64, 96, false>,
                            hipFuncAttributeMaxDynamicSharedMemorySize,
                            (int)lds_conv96);
  (void)hipFuncSetAttribute((const void*)&k_decode,
                            hipFuncAttributeMaxDynamicSharedMemorySize,
                            (int)lds_dec);

  k_z<<<1, 512, 0, stream>>>(zs, sp, pos_w, pos_b, zbuf);
  k_fmap<<<4096, 256, 0, stream>>>(zbuf, pe_w, pe_b, bufA);
  k_conv<64, 64, true><<<NSLOT * 64, 256, lds_conv64, stream>>>(bufA, c1_w, c1_b, bufB);
  k_conv<64, 64, true><<<NSLOT * 64, 256, lds_conv64, stream>>>(bufB, c2_w, c2_b, bufA);
  k_conv<64, 96, false><<<NSLOT * 64, 256, lds_conv96, stream>>>(bufA, c3_w, c3_b, bufT);
  k_decode<<<4096, 256, lds_dec, stream>>>(bufT, sc, sp, dw0, db0, dw1, db1,
                                           dnw, dnb, c0w, c0b, c1wc, c1bc,
                                           (float*)d_out);
  k_combine<<<256, 256, 0, stream>>>((float*)d_out);
}
